// LinearMultiHeadAttention_10342281249126
// MI455X (gfx1250) — matrix operand order repaired and checked by reading, not compiled, not run on hardware
//
#include <hip/hip_runtime.h>
#include <hip/hip_bf16.h>

// MI455X (gfx1250) wave32 WMMA implementation of linear multi-head attention.
// fp32->bf16 convert, 3 projection GEMMs (async-LDS double-buffered W tiles),
// flash attention (async-LDS double-buffered K/V staging), output GEMM w/ bias.

typedef __bf16 bf16_t;
typedef __attribute__((ext_vector_type(16))) __bf16 v16bf;
typedef __attribute__((ext_vector_type(8)))  __bf16 v8bf;
typedef __attribute__((ext_vector_type(8)))  float  v8f;

#define BB 4
#define LL 2048
#define EE 1024
#define HH 16
#define DD 64

__device__ __forceinline__ bf16_t f2bf(float f) {
  unsigned u = __float_as_uint(f);
  u = (u + 0x7FFFu + ((u >> 16) & 1u)) >> 16;   // round-to-nearest-even
  union { unsigned short s; bf16_t b; } cv;
  cv.s = (unsigned short)u;
  return cv.b;
}

__device__ __forceinline__ v16bf cat8(v8bf lo, v8bf hi) {
  return __builtin_shufflevector(lo, hi, 0,1,2,3,4,5,6,7,8,9,10,11,12,13,14,15);
}

__device__ __forceinline__ v8f wmma_bf16(v16bf a, v16bf b, v8f c) {
  return __builtin_amdgcn_wmma_f32_16x16x32_bf16(false, a, false, b, (short)0, c,
                                                 false, false);
}

// gfx1250 async global->LDS copy (ASYNCcnt-tracked). lds_off = byte offset into
// the wave's LDS allocation (low 32 bits of a flat LDS pointer).
__device__ __forceinline__ void async_copy_b128(unsigned lds_off, const void* src) {
  asm volatile("global_load_async_to_lds_b128 %0, %1, off"
               :: "v"(lds_off), "v"(src) : "memory");
}
__device__ __forceinline__ void wait_async0() {
  asm volatile("s_wait_asynccnt 0x0" ::: "memory");
}
__device__ __forceinline__ void wait_async1() {
  asm volatile("s_wait_asynccnt 0x1" ::: "memory");
}
__device__ __forceinline__ void wait_async2() {
  asm volatile("s_wait_asynccnt 0x2" ::: "memory");
}
__device__ __forceinline__ unsigned lds_off_of(const void* p) {
  return (unsigned)(size_t)p;   // flat LDS address truncates to LDS byte offset
}

__global__ void cvt_f32_to_bf16(const float* __restrict__ in,
                                bf16_t* __restrict__ out, int n) {
  int i = blockIdx.x * blockDim.x + threadIdx.x;
  if (i < n) out[i] = f2bf(in[i]);
}

// C[M,N] = A[M,K] * W[N,K]^T (torch Linear: x @ W.T).
// Block: 256 threads = 8 waves; each wave computes a 32x64 tile (8 WMMA/k-step).
// W tile (64 rows x 32 k) is double-buffered in LDS via async copies; the last
// k-step is peeled so the steady-state loop is branch-free.
__global__ __launch_bounds__(256)
void gemm_xwT(const bf16_t* __restrict__ A,   // [M,K] bf16 row-major
              const bf16_t* __restrict__ W,   // [N,K] bf16 row-major
              const float*  __restrict__ bias,// nullable, length N
              bf16_t* __restrict__ outb,      // bf16 out (used if outf==null)
              float*  __restrict__ outf,      // f32 out
              int M, int N, int K)
{
  __shared__ alignas(32) bf16_t wt[2][64 * 32];

  const int lane = threadIdx.x & 31;
  const int wave = threadIdx.x >> 5;
  const int m0 = blockIdx.x * 256 + wave * 32;
  const int n0 = blockIdx.y * 64;
  const int cn   = lane & 15;
  const int kg8  = (lane >> 4) * 8;
  const int kg16 = (lane >> 4) * 16;

  // Staging: 64 rows x 32 k = 256 x 16B segments, one per thread.
  const int srow = threadIdx.x >> 2;          // 0..63
  const int sseg = (threadIdx.x & 3) * 8;     // element offset 0,8,16,24
  const bf16_t* wsrc = W + (size_t)(n0 + srow) * K + sseg;
  const unsigned loff0 = lds_off_of(&wt[0][srow * 32 + sseg]);
  const unsigned loff1 = lds_off_of(&wt[1][srow * 32 + sseg]);

  v8f acc[8] = {};
  const bf16_t* arow0 = A + (size_t)(m0 + cn) * K;
  const bf16_t* arow1 = A + (size_t)(m0 + 16 + cn) * K;

  auto compute = [&](int buf, int k0) {
    const bf16_t* pa0 = arow0 + k0 + kg16;
    const bf16_t* pa1 = arow1 + k0 + kg16;
    v16bf af0 = cat8(*(const v8bf*)(pa0), *(const v8bf*)(pa0 + 8));
    v16bf af1 = cat8(*(const v8bf*)(pa1), *(const v8bf*)(pa1 + 8));
    const bf16_t* wb = &wt[buf][cn * 32 + kg16];
    v16bf b0 = *(const v16bf*)(wb);
    v16bf b1 = *(const v16bf*)(wb + 16 * 32);
    v16bf b2 = *(const v16bf*)(wb + 32 * 32);
    v16bf b3 = *(const v16bf*)(wb + 48 * 32);
    acc[0] = wmma_bf16(af0, b0, acc[0]);
    acc[1] = wmma_bf16(af0, b1, acc[1]);
    acc[2] = wmma_bf16(af0, b2, acc[2]);
    acc[3] = wmma_bf16(af0, b3, acc[3]);
    acc[4] = wmma_bf16(af1, b0, acc[4]);
    acc[5] = wmma_bf16(af1, b1, acc[5]);
    acc[6] = wmma_bf16(af1, b2, acc[6]);
    acc[7] = wmma_bf16(af1, b3, acc[7]);
  };

  async_copy_b128(loff0, wsrc);               // prologue: stage k0 = 0
  int k0 = 0;
  for (; k0 < K - 32; k0 += 32) {             // branch-free steady state
    const int buf = (k0 >> 5) & 1;
    async_copy_b128(buf ? loff0 : loff1, wsrc + k0 + 32);
    wait_async1();
    __syncthreads();
    compute(buf, k0);
    __syncthreads();
  }
  wait_async0();                              // peeled last k-step
  __syncthreads();
  compute((k0 >> 5) & 1, k0);

#pragma unroll
  for (int g = 0; g < 2; ++g) {
#pragma unroll
    for (int r = 0; r < 8; ++r) {
      int row = m0 + g * 16 + r + kg8;        // kg8 == (lane>>4)*8 row offset
      size_t base = (size_t)row * N + n0 + cn;
      if (outf) {
        outf[base + 0]  = acc[g * 4 + 0][r] + (bias ? bias[n0 + 0  + cn] : 0.f);
        outf[base + 16] = acc[g * 4 + 1][r] + (bias ? bias[n0 + 16 + cn] : 0.f);
        outf[base + 32] = acc[g * 4 + 2][r] + (bias ? bias[n0 + 32 + cn] : 0.f);
        outf[base + 48] = acc[g * 4 + 3][r] + (bias ? bias[n0 + 48 + cn] : 0.f);
      } else {
        outb[base + 0]  = f2bf(acc[g * 4 + 0][r]);
        outb[base + 16] = f2bf(acc[g * 4 + 1][r]);
        outb[base + 32] = f2bf(acc[g * 4 + 2][r]);
        outb[base + 48] = f2bf(acc[g * 4 + 3][r]);
      }
    }
  }
}

// Flash attention: grid = (L/64, H, B); block = 128 threads = 4 waves.
// Each wave owns 16 query rows; 32-key K/V chunks are double-buffered in LDS
// (K via async global->LDS, V transposed manually); last chunk is peeled.
__global__ __launch_bounds__(128)
void attn_fwd(const bf16_t* __restrict__ qp, const bf16_t* __restrict__ kp,
              const bf16_t* __restrict__ vp, bf16_t* __restrict__ ao)
{
  __shared__ alignas(32) bf16_t ktile[2][32 * 64];  // [key][d] row-major
  __shared__ alignas(32) bf16_t vtile[2][64 * 32];  // [d][key] transposed
  __shared__ alignas(32) bf16_t ptile[4][16 * 32];  // per-wave P scratch

  const int lane = threadIdx.x & 31;
  const int wave = threadIdx.x >> 5;
  const int b = blockIdx.z;
  const int h = blockIdx.y;
  const int q0 = blockIdx.x * 64 + wave * 16;
  const int cb   = lane & 15;
  const int kg8  = (lane >> 4) * 8;
  const int kg16 = (lane >> 4) * 16;
  const float scale = 0.03125f;  // E^-0.5 = 1024^-0.5

  const size_t head_off = ((size_t)b * LL) * EE + (size_t)h * DD;
  const bf16_t* kbase = kp + head_off;
  const bf16_t* vbase = vp + head_off;

  // Q fragments for this wave's 16 rows (D=64 -> two 16x32 A-frags)
  const bf16_t* pq = qp + head_off + (size_t)(q0 + cb) * EE + kg16;
  v16bf qf0 = cat8(*(const v8bf*)(pq),      *(const v8bf*)(pq + 8));
  v16bf qf1 = cat8(*(const v8bf*)(pq + 32), *(const v8bf*)(pq + 40));

  float m_run[8], s_run[8];
  v8f o0 = {}, o1 = {}, o2 = {}, o3 = {};
#pragma unroll
  for (int r = 0; r < 8; ++r) { m_run[r] = -1e30f; s_run[r] = 0.f; }

  // Stage chunk at key offset kc into buffer buf (2 async ops per thread).
  auto stage = [&](int kc, int buf) {
    bf16_t* kt = &ktile[buf][0];
    bf16_t* vt = &vtile[buf][0];
    for (int t = threadIdx.x; t < 256; t += 128) {
      int row = t >> 3, seg = (t & 7) * 8;
      async_copy_b128(lds_off_of(kt + row * 64 + seg),
                      kbase + (size_t)(kc + row) * EE + seg);
      v8bf vv = *(const v8bf*)(vbase + (size_t)(kc + row) * EE + seg);
#pragma unroll
      for (int j = 0; j < 8; ++j) vt[(seg + j) * 32 + row] = vv[j];
    }
  };

  auto process = [&](int buf) {
    const bf16_t* kt = &ktile[buf][0];
    const bf16_t* vt = &vtile[buf][0];

    // S = Q K^T for two 16-key subtiles (K-dim = D = 64 -> 2 WMMA each)
    v8f s0 = {}, s1 = {};
    {
      v16bf kf00 = *(const v16bf*)(kt + cb * 64 + kg16);
      v16bf kf01 = *(const v16bf*)(kt + cb * 64 + 32 + kg16);
      v16bf kf10 = *(const v16bf*)(kt + (16 + cb) * 64 + kg16);
      v16bf kf11 = *(const v16bf*)(kt + (16 + cb) * 64 + 32 + kg16);
      s0 = wmma_bf16(qf0, kf00, s0);
      s0 = wmma_bf16(qf1, kf01, s0);
      s1 = wmma_bf16(qf0, kf10, s1);
      s1 = wmma_bf16(qf1, kf11, s1);
    }

    // Online softmax; a row's 16 values live in one 16-lane half -> xor reduce.
    float corr[8], p0[8], p1[8];
#pragma unroll
    for (int r = 0; r < 8; ++r) {
      float a = s0[r] * scale;
      float c = s1[r] * scale;
      float mx = fmaxf(a, c);
      mx = fmaxf(mx, __shfl_xor(mx, 1));
      mx = fmaxf(mx, __shfl_xor(mx, 2));
      mx = fmaxf(mx, __shfl_xor(mx, 4));
      mx = fmaxf(mx, __shfl_xor(mx, 8));
      float mnew = fmaxf(m_run[r], mx);
      float cf = __expf(m_run[r] - mnew);
      p0[r] = __expf(a - mnew);
      p1[r] = __expf(c - mnew);
      float rs = p0[r] + p1[r];
      rs += __shfl_xor(rs, 1);
      rs += __shfl_xor(rs, 2);
      rs += __shfl_xor(rs, 4);
      rs += __shfl_xor(rs, 8);
      s_run[r] = s_run[r] * cf + rs;
      m_run[r] = mnew;
      corr[r] = cf;
    }
#pragma unroll
    for (int r = 0; r < 8; ++r) {
      o0[r] *= corr[r]; o1[r] *= corr[r]; o2[r] *= corr[r]; o3[r] *= corr[r];
    }

    // C-layout -> A-layout for P via per-wave LDS tile.
    bf16_t* pw = &ptile[wave][0];
#pragma unroll
    for (int r = 0; r < 8; ++r) {
      int row = r + kg8;
      pw[row * 32 + cb]      = f2bf(p0[r]);
      pw[row * 32 + 16 + cb] = f2bf(p1[r]);
    }
    __syncthreads();

    const bf16_t* pp = pw + (size_t)cb * 32 + kg16;
    v16bf pf = cat8(*(const v8bf*)pp, *(const v8bf*)(pp + 8));
    v16bf vf0 = *(const v16bf*)(vt + (0  + cb) * 32 + kg16);
    v16bf vf1 = *(const v16bf*)(vt + (16 + cb) * 32 + kg16);
    v16bf vf2 = *(const v16bf*)(vt + (32 + cb) * 32 + kg16);
    v16bf vf3 = *(const v16bf*)(vt + (48 + cb) * 32 + kg16);
    o0 = wmma_bf16(pf, vf0, o0);
    o1 = wmma_bf16(pf, vf1, o1);
    o2 = wmma_bf16(pf, vf2, o2);
    o3 = wmma_bf16(pf, vf3, o3);
  };

  stage(0, 0);                                // prologue
  int kc = 0;
  for (; kc < LL - 32; kc += 32) {            // branch-free steady state
    const int buf = (kc >> 5) & 1;
    stage(kc + 32, buf ^ 1);                  // prefetch next chunk
    wait_async2();                            // current chunk's 2 asyncs done
    __syncthreads();
    process(buf);
    __syncthreads();                          // readers done before re-stage
  }
  wait_async0();                              // peeled last chunk
  __syncthreads();
  process((kc >> 5) & 1);

  // Normalize and store [B,L,H*D] bf16 for the output projection.
#pragma unroll
  for (int r = 0; r < 8; ++r) {
    float inv = 1.0f / s_run[r];
    int row = q0 + r + kg8;
    bf16_t* po = ao + head_off + (size_t)row * EE;
    po[0  + cb] = f2bf(o0[r] * inv);
    po[16 + cb] = f2bf(o1[r] * inv);
    po[32 + cb] = f2bf(o2[r] * inv);
    po[48 + cb] = f2bf(o3[r] * inv);
  }
}

extern "C" void kernel_launch(void* const* d_in, const int* in_sizes, int n_in,
                              void* d_out, int out_size, void* d_ws, size_t ws_size,
                              hipStream_t stream) {
  (void)in_sizes; (void)n_in; (void)out_size; (void)ws_size;
  const float* q  = (const float*)d_in[0];
  const float* k  = (const float*)d_in[1];
  const float* v  = (const float*)d_in[2];
  const float* Wq = (const float*)d_in[3];
  const float* Wk = (const float*)d_in[4];
  const float* Wv = (const float*)d_in[5];
  const float* Wo = (const float*)d_in[6];
  const float* bo = (const float*)d_in[7];
  float* out = (float*)d_out;

  const size_t NE = (size_t)BB * LL * EE;   // 8,388,608 elements
  const size_t WE = (size_t)EE * EE;        // 1,048,576 elements
  bf16_t* p = (bf16_t*)d_ws;
  bf16_t* qb  = p; p += NE;
  bf16_t* kb  = p; p += NE;
  bf16_t* vb  = p; p += NE;
  bf16_t* wqb = p; p += WE;
  bf16_t* wkb = p; p += WE;
  bf16_t* wvb = p; p += WE;
  bf16_t* wob = p; p += WE;
  bf16_t* qpb = p; p += NE;
  bf16_t* kpb = p; p += NE;
  bf16_t* vpb = p; p += NE;
  bf16_t* ab  = p; p += NE;

  auto cvt = [&](const float* src, bf16_t* dst, size_t n) {
    cvt_f32_to_bf16<<<(unsigned)((n + 255) / 256), 256, 0, stream>>>(src, dst, (int)n);
  };
  cvt(q, qb, NE);  cvt(k, kb, NE);  cvt(v, vb, NE);
  cvt(Wq, wqb, WE); cvt(Wk, wkb, WE); cvt(Wv, wvb, WE); cvt(Wo, wob, WE);

  dim3 gg((BB * LL) / 256, EE / 64);  // 32 x 16
  gemm_xwT<<<gg, 256, 0, stream>>>(qb, wqb, nullptr, qpb, nullptr, BB * LL, EE, EE);
  gemm_xwT<<<gg, 256, 0, stream>>>(kb, wkb, nullptr, kpb, nullptr, BB * LL, EE, EE);
  gemm_xwT<<<gg, 256, 0, stream>>>(vb, wvb, nullptr, vpb, nullptr, BB * LL, EE, EE);

  dim3 ga(LL / 64, HH, BB);           // 32 x 16 x 4
  attn_fwd<<<ga, 128, 0, stream>>>(qpb, kpb, vpb, ab);

  gemm_xwT<<<gg, 256, 0, stream>>>(ab, wob, bo, nullptr, out, BB * LL, EE, EE);
}
